// MultiHeadSelfAttention_75883482186448
// MI455X (gfx1250) — matrix operand order repaired and checked by reading, not compiled, not run on hardware
//
#include <hip/hip_runtime.h>

// ---------------------------------------------------------------------------
// MHA for MI455X (gfx1250, wave32). Compute-bound (~60 GFLOP vs ~46MB HBM):
// all matmuls via v_wmma_f32_16x16x32_f16 (f32 accumulate). Flash-attention
// style (scores never hit memory). K/V tiles are staged into LDS with the
// CDNA5 async-to-LDS path (ASYNCcnt), double-buffered, shared by 4 waves.
// ---------------------------------------------------------------------------

typedef __attribute__((ext_vector_type(16))) _Float16 v16h;
typedef __attribute__((ext_vector_type(8)))  _Float16 v8h;
typedef __attribute__((ext_vector_type(8)))  float    v8f;

#define WMMA_F16(a, b, c) \
  __builtin_amdgcn_wmma_f32_16x16x32_f16(false, (a), false, (b), (short)0, (c), false, false)

// Problem constants.
#define NB   2
#define NS   2048
#define ND   1024
#define NH   16
#define NHD  64
#define N3D  3072
#define NM   (NB * NS)          // 4096 GEMM rows
#define SCALE 0.125f            // 64^-0.5

// Build a 16x32 f16 A-fragment for lane (row = lane%16, hi = lane/16):
// elems 0..7  = row[kbase + 16*hi .. +7]
// elems 8..15 = row[kbase + 16*hi + 8 .. +7]
static __device__ __forceinline__ v16h make_a_frag(const _Float16* __restrict__ row,
                                                   int kbase, int hi) {
  const v8h lo = *(const v8h*)(row + kbase + 16 * hi);
  const v8h up = *(const v8h*)(row + kbase + 16 * hi + 8);
  v16h r;
#pragma unroll
  for (int i = 0; i < 8; ++i) { r[i] = lo[i]; r[i + 8] = up[i]; }
  return r;
}

// CDNA5 async copy: 16B per lane, global -> LDS, tracked by ASYNCcnt.
static __device__ __forceinline__ void async_ld16(unsigned lds_off,
                                                  const _Float16* g) {
  asm volatile("global_load_async_to_lds_b128 %0, %1, off"
               :: "v"(lds_off), "v"(g) : "memory");
}
static __device__ __forceinline__ void wait_async0() {
  asm volatile("s_wait_asynccnt 0x0" ::: "memory");
}

// ---------------------------------------------------------------------------
// fp32 -> fp16 flat convert (src)
__global__ __launch_bounds__(256) void cvt_f16_kernel(const float* __restrict__ x,
                                                      _Float16* __restrict__ y, int n) {
  int i = blockIdx.x * 256 + threadIdx.x;
  if (i < n) y[i] = (_Float16)x[i];
}

// W [D,3D] fp32 -> Wt [3D,D] fp16 (transposed so B-fragments read contiguous rows)
__global__ __launch_bounds__(256) void wt_kernel(const float* __restrict__ w,
                                                 _Float16* __restrict__ wt) {
  int i = blockIdx.x * 256 + threadIdx.x;   // i in [0, 3072*1024)
  int n = i >> 10;                          // output column (0..3071)
  int k = i & 1023;                         // reduction index
  wt[i] = (_Float16)w[(size_t)k * N3D + n];
}

// ---------------------------------------------------------------------------
// QKV projection: C[4096,3072] = srch @ W + b, scattered into Q/K/Vt (f16).
// One wave computes a 16(M) x 64(N) tile: 4 accumulators, K-loop of 32 steps.
__global__ __launch_bounds__(128) void qkv_gemm_kernel(
    const _Float16* __restrict__ srch,   // [4096,1024]
    const _Float16* __restrict__ wt,     // [3072,1024]  (W^T)
    const float*    __restrict__ bias,   // [3072]
    _Float16* __restrict__ Qh,           // [B,H,S,64]
    _Float16* __restrict__ Kh,           // [B,H,S,64]
    _Float16* __restrict__ Vt) {         // [B,H,64,S]
  const int lane = threadIdx.x & 31;
  const int wave = threadIdx.x >> 5;
  const int n16  = lane & 15;
  const int hi   = lane >> 4;
  const int m0   = blockIdx.y * 64 + wave * 16;
  const int n0   = blockIdx.x * 64;

  v8f acc[4] = {};
  const _Float16* arow = srch + (size_t)(m0 + n16) * ND;

#pragma unroll 1
  for (int kk = 0; kk < ND / 32; ++kk) {
    const int kb = kk * 32;
    const v16h a = make_a_frag(arow, kb, hi);
#pragma unroll
    for (int t = 0; t < 4; ++t) {
      const v16h b = *(const v16h*)(wt + (size_t)(n0 + t * 16 + n16) * ND + kb + 16 * hi);
      acc[t] = WMMA_F16(a, b, acc[t]);
    }
  }

  // Epilogue: +bias, convert, scatter into Q / K / V^T.
#pragma unroll
  for (int t = 0; t < 4; ++t) {
    const int col    = n0 + t * 16 + n16;
    const float bv   = bias[col];
    const int part   = col >> 10;     // 0=Q 1=K 2=V (wave-uniform: n0 % 64 == 0)
    const int within = col & 1023;
    const int h      = within >> 6;
    const int d      = within & 63;
#pragma unroll
    for (int i = 0; i < 8; ++i) {
      const int row = m0 + i + 8 * hi;       // global (b*S + s)
      const int b_  = row >> 11;
      const int s   = row & 2047;
      const _Float16 v = (_Float16)(acc[t][i] + bv);
      const size_t bh = (size_t)(b_ * NH + h);
      if (part == 0)      Qh[(bh * NS + s) * NHD + d] = v;
      else if (part == 1) Kh[(bh * NS + s) * NHD + d] = v;
      else                Vt[(bh * NHD + d) * NS + s] = v;
    }
  }
}

// ---------------------------------------------------------------------------
// Flash attention. Block = 4 waves sharing one (b,h); each wave owns a 16-row
// Q tile. Per 32-key block: K tile (32x64) and V^T tile (64x32) are async-
// copied into LDS (double-buffered; 4 async b128 per wave per tile-pair),
// then 4 WMMAs for S = Q K^T, online softmax (shfl_xor inside 16-lane
// fragment halves), P -> LDS -> A-fragment, 4 WMMAs for O += P V.
__global__ __launch_bounds__(128) void flash_attn_kernel(
    const _Float16* __restrict__ Qh,   // [B,H,S,64]
    const _Float16* __restrict__ Kh,   // [B,H,S,64]
    const _Float16* __restrict__ Vt,   // [B,H,64,S]
    float* __restrict__ out) {         // [B,S,D]
  __shared__ alignas(32) _Float16 lds_k[2][32 * 64];   // [key][d]
  __shared__ alignas(32) _Float16 lds_v[2][64 * 32];   // [d][key]
  __shared__ alignas(32) _Float16 lds_p[4][16 * 32];   // per-wave P bounce

  const int tid  = threadIdx.x;
  const int lane = tid & 31;
  const int wave = tid >> 5;
  const int n16  = lane & 15;
  const int hi   = lane >> 4;
  const int h    = blockIdx.y;
  const int b    = blockIdx.z;
  const int bh   = b * NH + h;
  const int q0   = blockIdx.x * 64 + wave * 16;

  const unsigned kls[2] = { (unsigned)(size_t)&lds_k[0][0], (unsigned)(size_t)&lds_k[1][0] };
  const unsigned vls[2] = { (unsigned)(size_t)&lds_v[0][0], (unsigned)(size_t)&lds_v[1][0] };

  // Issue async tile-pair load for key block jn into buffer bf.
  // K tile: 4KB contiguous in Kh. V tile: 64 rows x 64B from V^T.
  auto issue_tile = [&](int jn, int bf) {
    const _Float16* gk = Kh + ((size_t)bh * NS + jn) * NHD;   // 2048 halfs, contiguous
    async_ld16(kls[bf] + (unsigned)tid * 16u,         gk + tid * 8);
    async_ld16(kls[bf] + (unsigned)(tid + 128) * 16u, gk + (tid + 128) * 8);
#pragma unroll
    for (int half = 0; half < 2; ++half) {
      const int c  = tid + half * 128;      // chunk 0..255
      const int d  = c >> 2;
      const int kp = c & 3;
      async_ld16(vls[bf] + (unsigned)(d * 32 + kp * 8) * 2u,
                 Vt + ((size_t)bh * NHD + d) * NS + jn + kp * 8);
    }
  };

  // Q tile (16 x 64) as two A-fragments, loaded once.
  const _Float16* qrow = Qh + ((size_t)bh * NS + q0 + n16) * NHD;
  const v16h aq0 = make_a_frag(qrow, 0, hi);
  const v16h aq1 = make_a_frag(qrow, 32, hi);

  v8f  o[4] = {};
  float m[8], l[8];
#pragma unroll
  for (int i = 0; i < 8; ++i) { m[i] = -3.0e38f; l[i] = 0.f; }

  _Float16* plw = lds_p[wave];

  issue_tile(0, 0);

  for (int j = 0; j < NS; j += 32) {
    const int cur = (j >> 5) & 1;
    // Our async writes landed; barrier publishes every wave's portion and
    // guarantees the other buffer is no longer read before it is overwritten.
    wait_async0();
    __syncthreads();
    if (j + 32 < NS) issue_tile(j + 32, 1 - cur);

    const _Float16* kt = lds_k[cur];
    const _Float16* vt = lds_v[cur];

    // ---- scores S[16 x 32] = Q @ K^T (two 16-col fragments), from LDS ----
    v8f s0 = {}, s1 = {};
    {
      v16h bk = *(const v16h*)(kt + (size_t)n16 * NHD + 16 * hi);
      s0 = WMMA_F16(aq0, bk, s0);
      bk = *(const v16h*)(kt + (size_t)n16 * NHD + 32 + 16 * hi);
      s0 = WMMA_F16(aq1, bk, s0);
      bk = *(const v16h*)(kt + (size_t)(16 + n16) * NHD + 16 * hi);
      s1 = WMMA_F16(aq0, bk, s1);
      bk = *(const v16h*)(kt + (size_t)(16 + n16) * NHD + 32 + 16 * hi);
      s1 = WMMA_F16(aq1, bk, s1);
    }

    // ---- online softmax; element i lives in row (i + 8*hi), col n16 ----
#pragma unroll
    for (int i = 0; i < 8; ++i) {
      const float x0 = s0[i] * SCALE;
      const float x1 = s1[i] * SCALE;
      float mb = fmaxf(x0, x1);
      mb = fmaxf(mb, __shfl_xor(mb, 1, 32));
      mb = fmaxf(mb, __shfl_xor(mb, 2, 32));
      mb = fmaxf(mb, __shfl_xor(mb, 4, 32));
      mb = fmaxf(mb, __shfl_xor(mb, 8, 32));
      const float mn   = fmaxf(m[i], mb);
      const float corr = __expf(m[i] - mn);
      m[i] = mn;
      const float p0 = __expf(x0 - mn);
      const float p1 = __expf(x1 - mn);
      float rs = p0 + p1;
      rs += __shfl_xor(rs, 1, 32);
      rs += __shfl_xor(rs, 2, 32);
      rs += __shfl_xor(rs, 4, 32);
      rs += __shfl_xor(rs, 8, 32);
      l[i] = l[i] * corr + rs;
#pragma unroll
      for (int t = 0; t < 4; ++t) o[t][i] *= corr;
      // stash P (C-layout -> LDS row-major [row][key]); per-wave region
      const int row = i + 8 * hi;
      plw[row * 32 + n16]      = (_Float16)p0;
      plw[row * 32 + 16 + n16] = (_Float16)p1;
    }

    // per-wave DS ordering: stores above must land before the A-frag loads
    asm volatile("s_wait_dscnt 0x0" ::: "memory");
    v16h ap;
    {
      const _Float16* prow = plw + n16 * 32;
      const v8h lo = *(const v8h*)(prow + 16 * hi);
      const v8h up = *(const v8h*)(prow + 16 * hi + 8);
#pragma unroll
      for (int i = 0; i < 8; ++i) { ap[i] = lo[i]; ap[i + 8] = up[i]; }
    }
    asm volatile("" ::: "memory");

    // ---- O[16 x 64] += P @ V  (B-operand = contiguous rows of V^T tile) ----
#pragma unroll
    for (int t = 0; t < 4; ++t) {
      const v16h bv = *(const v16h*)(vt + (size_t)(t * 16 + n16) * 32 + 16 * hi);
      o[t] = WMMA_F16(ap, bv, o[t]);
    }
  }

  // ---- normalize and write fp32 output: out[b, s, h*64 + d] ----
  float* obase = out + (size_t)b * NS * ND + (size_t)h * NHD;
#pragma unroll
  for (int t = 0; t < 4; ++t) {
    const int d = t * 16 + n16;
#pragma unroll
    for (int i = 0; i < 8; ++i) {
      const int row = q0 + i + 8 * hi;
      obase[(size_t)row * ND + d] = o[t][i] / l[i];
    }
  }
}

// ---------------------------------------------------------------------------
extern "C" void kernel_launch(void* const* d_in, const int* in_sizes, int n_in,
                              void* d_out, int out_size, void* d_ws, size_t ws_size,
                              hipStream_t stream) {
  (void)in_sizes; (void)n_in; (void)out_size; (void)ws_size;
  const float* src  = (const float*)d_in[0];   // [B,S,D]
  const float* wqkv = (const float*)d_in[1];   // [D,3D]
  const float* bqkv = (const float*)d_in[2];   // [3D]
  float* out = (float*)d_out;                  // [B,S,D]

  // Workspace layout (all f16): src 8MB | Wt 6MB | Q 8MB | K 8MB | Vt 8MB
  char* ws = (char*)d_ws;
  const size_t sz_src = (size_t)NM * ND * 2;             // 8388608
  const size_t sz_wt  = (size_t)N3D * ND * 2;            // 6291456
  const size_t sz_qkv = (size_t)NB * NH * NS * NHD * 2;  // 8388608 each
  _Float16* srch = (_Float16*)ws;
  _Float16* Wt   = (_Float16*)(ws + sz_src);
  _Float16* Qh   = (_Float16*)(ws + sz_src + sz_wt);
  _Float16* Kh   = (_Float16*)(ws + sz_src + sz_wt + sz_qkv);
  _Float16* Vt   = (_Float16*)(ws + sz_src + sz_wt + 2 * sz_qkv);

  const int n_src = NM * ND;   // 4194304
  const int n_wt  = N3D * ND;  // 3145728
  cvt_f16_kernel<<<(n_src + 255) / 256, 256, 0, stream>>>(src, srch, n_src);
  wt_kernel<<<(n_wt + 255) / 256, 256, 0, stream>>>(wqkv, Wt);
  qkv_gemm_kernel<<<dim3(N3D / 64, NM / 64), 128, 0, stream>>>(srch, Wt, bqkv, Qh, Kh, Vt);
  flash_attn_kernel<<<dim3(NS / 64, NH, NB), 128, 0, stream>>>(Qh, Kh, Vt, out);
}
